// RPN_70738111365912
// MI455X (gfx1250) — matrix operand order repaired and checked by reading, not compiled, not run on hardware
//
#include <hip/hip_runtime.h>
#include <math.h>

// ---------------- problem constants ----------------
#define BATCH 64
#define NPTS  2048
#define FDIM  128
#define PPROP 64
#define KNNK  8
#define SSAMP 8
#define MROWS (PPROP * SSAMP)   // 512 sample points per batch
#define NEDGE (MROWS * KNNK)    // 4096 edge columns per batch

typedef __attribute__((ext_vector_type(16))) _Float16 v16h;
typedef __attribute__((ext_vector_type(8)))  _Float16 v8h;
typedef __attribute__((ext_vector_type(8)))  float    v8f;

__device__ __forceinline__ float sigmoidf_(float x) { return 1.0f / (1.0f + expf(-x)); }

// =====================================================================
// Tiled WMMA GEMM:  Y[b,o,n] = act( W[o,:] . X[b,:,n] + bias[o] )
//   act: if BN: relu(y*s[o]+t[o]) else identity
//   Block: 128 threads (4 waves). Output tile: (MT*16) x 64 per block.
//   K staged in steps of 32 through double-buffered LDS (f32 -> f16 on load),
//   so global loads for chunk k+1 overlap the WMMAs of chunk k.
// =====================================================================
template <int MT, bool BN>
__global__ __launch_bounds__(128) void gemm_wmma(
    const float* __restrict__ X, const float* __restrict__ W,
    const float* __restrict__ bias, const float* __restrict__ scale,
    const float* __restrict__ shift, float* __restrict__ Y,
    int Cin, int Cout, int Nn)
{
    constexpr int LDW = 40;              // halves per LDS row (16B aligned, bank-spread)
    constexpr int UT  = (MT + 3) / 4;    // M-tiles per wave

    __shared__ _Float16 lds_w[2][MT * 16 * LDW];
    __shared__ _Float16 lds_x[2][64 * LDW];

    const int b    = blockIdx.y;
    const int n0   = blockIdx.x * 64;
    const int tid  = threadIdx.x;
    // wave index is uniform per wave: force scalar so WMMA guards become
    // s_cbranch (EXEC stays all-1s around every v_wmma, per ISA requirement)
    const int wave = __builtin_amdgcn_readfirstlane(tid >> 5);
    const int lane = tid & 31;
    const int lrow = lane & 15;
    const int kA   = (lane < 16) ? 0 : 8;    // A-matrix K base (ISA 16-bit A layout)
    const int kB   = (lane < 16) ? 0 : 8;    // B-matrix K base (same K order as A)

    const float* Xb = X + (size_t)b * Cin * Nn;

    // stage one 32-wide K chunk (W rows x 32 and X^T 64 x 32) into LDS buffer `buf`
    auto stage = [&](int buf, int k0) {
        for (int idx = tid; idx < MT * 16 * 32; idx += 128) {
            int o = idx >> 5, i = idx & 31;
            float v = 0.0f;
            if (o < Cout && (k0 + i) < Cin) v = W[(size_t)o * Cin + k0 + i];
            lds_w[buf][o * LDW + i] = (_Float16)v;
        }
        for (int idx = tid; idx < 32 * 64; idx += 128) {
            int i = idx >> 6, n = idx & 63;
            float v = ((k0 + i) < Cin) ? Xb[(size_t)(k0 + i) * Nn + n0 + n] : 0.0f;
            lds_x[buf][n * LDW + i] = (_Float16)v;
        }
    };

    v8f acc[UT][4];
    for (int u = 0; u < UT; ++u)
        for (int nt = 0; nt < 4; ++nt)
            for (int e = 0; e < 8; ++e) acc[u][nt][e] = 0.0f;

    stage(0, 0);
    __syncthreads();

    int cur = 0;
    for (int k0 = 0; k0 < Cin; k0 += 32) {
        // prefetch the chunk after next (gfx1250 global_prefetch_b8)
        if (k0 + 64 < Cin) {
            __builtin_prefetch(&Xb[(size_t)(k0 + 64 + (tid >> 3)) * Nn + n0 + (tid & 7) * 8], 0, 1);
            __builtin_prefetch(&W[(size_t)(tid) * Cin + k0 + 64], 0, 1);
        }
        // stage next chunk into the other buffer while computing this one
        if (k0 + 32 < Cin) stage(cur ^ 1, k0 + 32);

        // B fragments: lane holds column n = nt*16+lrow, K = {kB..kB+7, kB+16..kB+23}
        v16h bfrag[4];
#pragma unroll
        for (int nt = 0; nt < 4; ++nt) {
            const _Float16* p = &lds_x[cur][(nt * 16 + lrow) * LDW + kB];
            v8h lo = *(const v8h*)p;
            v8h hi = *(const v8h*)(p + 16);
            bfrag[nt] = __builtin_shufflevector(lo, hi, 0,1,2,3,4,5,6,7,8,9,10,11,12,13,14,15);
        }
#pragma unroll
        for (int u = 0; u < UT; ++u) {
            int tm = wave + 4 * u;
            if (tm < MT) {
                // A fragment: row o = tm*16+lrow, K = {kA..kA+7, kA+16..kA+23}
                const _Float16* p = &lds_w[cur][(tm * 16 + lrow) * LDW + kA];
                v8h lo = *(const v8h*)p;
                v8h hi = *(const v8h*)(p + 16);
                v16h afrag = __builtin_shufflevector(lo, hi, 0,1,2,3,4,5,6,7,8,9,10,11,12,13,14,15);
#pragma unroll
                for (int nt = 0; nt < 4; ++nt)
                    acc[u][nt] = __builtin_amdgcn_wmma_f32_16x16x32_f16(
                        false, afrag, false, bfrag[nt], (short)0, acc[u][nt], false, false);
            }
        }
        __syncthreads();
        cur ^= 1;
    }

    // epilogue: bias (+ BN affine + relu), bounds-checked on Cout
    float* Yb = Y + (size_t)b * Cout * Nn;
#pragma unroll
    for (int u = 0; u < UT; ++u) {
        int tm = wave + 4 * u;
        if (tm < MT) {
#pragma unroll
            for (int nt = 0; nt < 4; ++nt) {
                int n = n0 + nt * 16 + lrow;
#pragma unroll
                for (int j = 0; j < 8; ++j) {
                    int o = tm * 16 + j + ((lane >= 16) ? 8 : 0);
                    if (o < Cout) {
                        float v = acc[u][nt][j] + bias[o];
                        if (BN) {
                            v = v * scale[o] + shift[o];
                            v = v > 0.0f ? v : 0.0f;
                        }
                        Yb[(size_t)o * Nn + n] = v;
                    }
                }
            }
        }
    }
}

template <int MT, bool BN>
static void run_gemm(const float* X, const float* W, const float* bias,
                     const float* s, const float* t, float* Y,
                     int Cin, int Cout, int Nn, hipStream_t st)
{
    dim3 g(Nn / 64, BATCH), blk(128);
    gemm_wmma<MT, BN><<<g, blk, 0, st>>>(X, W, bias, s, t, Y, Cin, Cout, Nn);
}

// =====================================================================
// Elementwise / gather / reduction kernels
// =====================================================================
__global__ void k_add(const float* __restrict__ a, const float* __restrict__ c,
                      float* __restrict__ o, long n) {
    long i = (long)blockIdx.x * blockDim.x + threadIdx.x;
    if (i < n) o[i] = a[i] + c[i];
}

// featxyz (B,131,N): rows 0..127 = feat, rows 128..130 = xyz^T
__global__ void k_featxyz(const float* __restrict__ feat, const float* __restrict__ xyz,
                          float* __restrict__ out) {
    long i = (long)blockIdx.x * blockDim.x + threadIdx.x;
    long tot = (long)BATCH * 131 * NPTS;
    if (i >= tot) return;
    int b = i / (131 * NPTS);
    int r = (i / NPTS) % 131;
    int n = i % NPTS;
    out[i] = (r < 128) ? feat[((size_t)b * 128 + r) * NPTS + n]
                       : xyz[((size_t)b * NPTS + n) * 3 + (r - 128)];
}

// feat += off[:, 0:128]   (off stored as (B,132,N))
__global__ void k_featadd(float* __restrict__ feat, const float* __restrict__ off) {
    long i = (long)blockIdx.x * blockDim.x + threadIdx.x;
    long tot = (long)BATCH * 128 * NPTS;
    if (i >= tot) return;
    int b = i / (128 * NPTS);
    int r = (i / NPTS) % 128;
    int n = i % NPTS;
    feat[i] += off[((size_t)b * 132 + r) * NPTS + n];
}

// center_pred = off[128:131]^T + xyz ; objectness_pred = off[131]
__global__ void k_centerobj(const float* __restrict__ off, const float* __restrict__ xyz,
                            float* __restrict__ center, float* __restrict__ obj) {
    long i = (long)blockIdx.x * blockDim.x + threadIdx.x;
    long tot = (long)BATCH * NPTS;
    if (i >= tot) return;
    int b = i / NPTS, n = i % NPTS;
    const float* ob = off + (size_t)b * 132 * NPTS;
    for (int d = 0; d < 3; ++d)
        center[i * 3 + d] = ob[(size_t)(128 + d) * NPTS + n] + xyz[i * 3 + d];
    obj[i] = ob[(size_t)131 * NPTS + n];
}

// feat_s (B,129,N): row0 = sigmoid(mask_pred), rows 1..128 = feat
__global__ void k_feats(const float* __restrict__ maskraw, const float* __restrict__ feat,
                        float* __restrict__ out) {
    long i = (long)blockIdx.x * blockDim.x + threadIdx.x;
    long tot = (long)BATCH * 129 * NPTS;
    if (i >= tot) return;
    int b = i / (129 * NPTS);
    int r = (i / NPTS) % 129;
    int n = i % NPTS;
    out[i] = (r == 0) ? sigmoidf_(maskraw[(size_t)b * NPTS + n])
                      : feat[((size_t)b * 128 + (r - 1)) * NPTS + n];
}

// Farthest point sampling: one block per batch (sequential 63 argmax steps)
__global__ __launch_bounds__(256) void k_fps(
    const float* __restrict__ center, float* __restrict__ prop_xyz,
    int* __restrict__ fps_idx, const float* __restrict__ obj_raw,
    float* __restrict__ score)
{
    __shared__ float mind[NPTS];
    __shared__ float rv[256];
    __shared__ int   ri[256];
    int b = blockIdx.x, tid = threadIdx.x;
    const float* pts = center + (size_t)b * NPTS * 3;
    for (int n = tid; n < NPTS; n += 256) mind[n] = 3.4e38f;
    if (tid < 3) prop_xyz[((size_t)b * PPROP) * 3 + tid] = pts[tid];
    if (tid == 0) {
        fps_idx[b * PPROP] = 0;
        score[b * PPROP]   = sigmoidf_(obj_raw[(size_t)b * NPTS]);
    }
    __syncthreads();
    int last = 0;
    for (int p = 1; p < PPROP; ++p) {
        float lx = pts[last * 3], ly = pts[last * 3 + 1], lz = pts[last * 3 + 2];
        float bestv = -1.0f; int besti = NPTS;
        for (int n = tid; n < NPTS; n += 256) {
            float dx = pts[n * 3] - lx, dy = pts[n * 3 + 1] - ly, dz = pts[n * 3 + 2] - lz;
            float d = dx * dx + dy * dy + dz * dz;
            float m = mind[n];
            m = d < m ? d : m;
            mind[n] = m;
            if (m > bestv || (m == bestv && n < besti)) { bestv = m; besti = n; }
        }
        rv[tid] = bestv; ri[tid] = besti;
        __syncthreads();
        for (int off = 128; off > 0; off >>= 1) {
            if (tid < off) {
                if (rv[tid + off] > rv[tid] ||
                    (rv[tid + off] == rv[tid] && ri[tid + off] < ri[tid])) {
                    rv[tid] = rv[tid + off]; ri[tid] = ri[tid + off];
                }
            }
            __syncthreads();
        }
        last = ri[0];
        if (tid < 3) prop_xyz[((size_t)b * PPROP + p) * 3 + tid] = pts[last * 3 + tid];
        if (tid == 0) {
            fps_idx[b * PPROP + p] = last;
            score[b * PPROP + p]   = sigmoidf_(obj_raw[(size_t)b * NPTS + last]);
        }
        __syncthreads();
    }
}

// sample_xyz (B,512,3) = proposal_xyz + proto*lwh
__global__ void k_sample(const float* __restrict__ propxyz, const float* __restrict__ lwh,
                         float* __restrict__ sample) {
    long i = (long)blockIdx.x * blockDim.x + threadIdx.x;
    long tot = (long)BATCH * MROWS;
    if (i >= tot) return;
    int b = i / MROWS, m = i % MROWS;
    int p = m / SSAMP, s = m % SSAMP;
    float pr0 = ((s >> 2) & 1) ? 0.25f : -0.25f;
    float pr1 = ((s >> 1) & 1) ? 0.25f : -0.25f;
    float pr2 = (s & 1) ? 0.25f : -0.25f;
    const float* q = propxyz + ((size_t)b * PPROP + p) * 3;
    sample[i * 3 + 0] = q[0] + pr0 * lwh[b * 3 + 0];
    sample[i * 3 + 1] = q[1] + pr1 * lwh[b * 3 + 1];
    sample[i * 3 + 2] = q[2] + pr2 * lwh[b * 3 + 2];
}

// K=8 nearest neighbors by insertion sort (matches top_k(-d) order)
__global__ void k_knn(const float* __restrict__ sample, const float* __restrict__ xyz,
                      int* __restrict__ knn) {
    long i = (long)blockIdx.x * blockDim.x + threadIdx.x;
    long tot = (long)BATCH * MROWS;
    if (i >= tot) return;
    int b = i / MROWS;
    float sx = sample[i * 3], sy = sample[i * 3 + 1], sz = sample[i * 3 + 2];
    float bd[KNNK]; int bi[KNNK];
    for (int k = 0; k < KNNK; ++k) { bd[k] = 3.4e38f; bi[k] = 0; }
    const float* xb = xyz + (size_t)b * NPTS * 3;
    for (int n = 0; n < NPTS; ++n) {
        float dx = xb[n * 3] - sx, dy = xb[n * 3 + 1] - sy, dz = xb[n * 3 + 2] - sz;
        float d = dx * dx + dy * dy + dz * dz;
        if (d < bd[KNNK - 1]) {
            int k = KNNK - 1;
            while (k > 0 && d < bd[k - 1]) { bd[k] = bd[k - 1]; bi[k] = bi[k - 1]; --k; }
            bd[k] = d; bi[k] = n;
        }
    }
    for (int k = 0; k < KNNK; ++k) knn[i * KNNK + k] = bi[k];
}

// edge (B,134,4096): rows 0-2 = knn_xyz - q, 3-5 = q, 6..133 = gathered pre_feat
__global__ void k_edge(const int* __restrict__ knn, const float* __restrict__ xyz,
                       const float* __restrict__ sample, const float* __restrict__ prefeat,
                       float* __restrict__ edge) {
    long i = (long)blockIdx.x * blockDim.x + threadIdx.x;
    long tot = (long)BATCH * NEDGE;
    if (i >= tot) return;
    int b = i / NEDGE;
    int rem = i % NEDGE;
    int m = rem / KNNK;
    int idx = knn[i];
    const float* q  = sample + ((size_t)b * MROWS + m) * 3;
    const float* px = xyz + ((size_t)b * NPTS + idx) * 3;
    float* e = edge + (size_t)b * 134 * NEDGE + rem;
    for (int d = 0; d < 3; ++d) e[(size_t)d * NEDGE] = px[d] - q[d];
    for (int d = 0; d < 3; ++d) e[(size_t)(3 + d) * NEDGE] = q[d];
    const float* pf = prefeat + (size_t)b * 128 * NPTS + idx;
    for (int f = 0; f < 128; ++f) e[(size_t)(6 + f) * NEDGE] = pf[(size_t)f * NPTS];
}

// max over K=8 + psf reshape: psf[b, f*8+s, p] = max_k nf[b,f,(p*8+s)*8+k]
__global__ void k_maxpsf(const float* __restrict__ y, float* __restrict__ psf) {
    long i = (long)blockIdx.x * blockDim.x + threadIdx.x;
    long tot = (long)BATCH * 128 * MROWS;
    if (i >= tot) return;
    int b = i / (128 * MROWS);
    int rem = i % (128 * MROWS);
    int f = rem / MROWS, m = rem % MROWS;
    int p = m / SSAMP, s = m % SSAMP;
    const float* yp = y + ((size_t)b * 128 + f) * NEDGE + (size_t)m * KNNK;
    float v = yp[0];
    for (int k = 1; k < KNNK; ++k) v = yp[k] > v ? yp[k] : v;
    psf[((size_t)b * 1024 + f * SSAMP + s) * PPROP + p] = v;
}

// proposal_feat += center_emb(score):  pf[b,f,p] += Wce[f]*score[b,p] + bce[f]
__global__ void k_ceadd(float* __restrict__ pf, const float* __restrict__ Wce,
                        const float* __restrict__ bce, const float* __restrict__ score) {
    long i = (long)blockIdx.x * blockDim.x + threadIdx.x;
    long tot = (long)BATCH * 128 * PPROP;
    if (i >= tot) return;
    int b = i / (128 * PPROP);
    int rem = i % (128 * PPROP);
    int f = rem / PPROP, p = rem % PPROP;
    pf[i] += Wce[f] * score[b * PPROP + p] + bce[f];
}

// bboxes: first 3 channels += proposal_xyz
__global__ void k_bbox(const float* __restrict__ prop, const float* __restrict__ propxyz,
                       float* __restrict__ bb) {
    long i = (long)blockIdx.x * blockDim.x + threadIdx.x;
    long tot = (long)BATCH * PPROP;
    if (i >= tot) return;
    int b = i / PPROP, p = i % PPROP;
    for (int d = 0; d < 5; ++d) {
        float v = prop[((size_t)b * 5 + d) * PPROP + p];
        bb[i * 5 + d] = (d < 3) ? v + propxyz[i * 3 + d] : v;
    }
}

// =====================================================================
// Launch
// =====================================================================
static inline dim3 grd(long n) { return dim3((unsigned)((n + 255) / 256)); }

extern "C" void kernel_launch(void* const* d_in, const int* in_sizes, int n_in,
                              void* d_out, int out_size, void* d_ws, size_t ws_size,
                              hipStream_t stream)
{
    const float** in = (const float**)d_in;
    // input order: xyz, geo_feat, mask_feat, lwh, then params in insertion order
    const float* xyz  = in[0];
    const float* geo  = in[1];
    const float* msk  = in[2];
    const float* lwh  = in[3];
    // fc_mask: 4..13 | fc_center: 14..23 | center_emb: 24,25 | pre_mlp: 26..31
    // shared_mlp: 32..39 | aggr_conv: 40..43 | fc_proposal: 44..53

    float* out = (float*)d_out;
    const long BN_ = (long)BATCH * NPTS;
    float* o_mask    = out;                     // (B,N)
    float* o_obj     = out + BN_;               // (B,N)
    float* o_center  = out + 2 * BN_;           // (B,N,3)
    float* o_bbox    = out + 5 * BN_;           // (B,P,5)
    float* o_propxyz = out + 5 * BN_ + (long)BATCH * PPROP * 5; // (B,P,3)

    // workspace bump allocator
    float* wsf = (float*)d_ws;
    size_t off = 0;
    auto alloc = [&](size_t n) { float* p = wsf + off; off += n; return p; };
    float* S0  = alloc((size_t)BATCH * 128 * NPTS);   // feat
    float* S1  = alloc((size_t)BATCH * 144 * NPTS);   // ping (up to 144 rows)
    float* S2  = alloc((size_t)BATCH * 144 * NPTS);   // pong
    float* E0  = alloc((size_t)BATCH * 134 * NEDGE);  // edge / shared_mlp out
    float* E1  = alloc((size_t)BATCH * 128 * NEDGE);
    float* PSF = alloc((size_t)BATCH * 1024 * PPROP);
    float* PF0 = alloc((size_t)BATCH * 128 * PPROP);
    float* PF1 = alloc((size_t)BATCH * 128 * PPROP);
    float* PF2 = alloc((size_t)BATCH * 5 * PPROP);
    float* SCORE  = alloc((size_t)BATCH * PPROP);
    float* SAMPLE = alloc((size_t)BATCH * MROWS * 3);
    int* FPSIDX = (int*)(wsf + off); off += (size_t)BATCH * PPROP;
    int* KNN    = (int*)(wsf + off); off += (size_t)BATCH * MROWS * KNNK;

    // 1) feat = geo + mask
    k_add<<<grd((long)BATCH * 128 * NPTS), 256, 0, stream>>>(geo, msk, S0, (long)BATCH * 128 * NPTS);

    // 2) fc_mask: 128->128 bn, 128->128 bn, 128->1
    run_gemm<8, true >(S0, in[4],  in[5],  in[6],  in[7],  S1, 128, 128, NPTS, stream);
    run_gemm<8, true >(S1, in[8],  in[9],  in[10], in[11], S2, 128, 128, NPTS, stream);
    run_gemm<1, false>(S2, in[12], in[13], nullptr, nullptr, o_mask, 128, 1, NPTS, stream);

    // 3) fc_center: concat(feat, xyz^T) -> 131->128 bn, 128->128 bn, 128->132
    k_featxyz<<<grd((long)BATCH * 131 * NPTS), 256, 0, stream>>>(S0, xyz, S1);
    run_gemm<8, true >(S1, in[14], in[15], in[16], in[17], S2, 131, 128, NPTS, stream);
    run_gemm<8, true >(S2, in[18], in[19], in[20], in[21], S1, 128, 128, NPTS, stream);
    run_gemm<9, false>(S1, in[22], in[23], nullptr, nullptr, S2, 128, 132, NPTS, stream);

    // 4) feat += off[:128]; center_pred / objectness_pred outputs
    k_featadd<<<grd((long)BATCH * 128 * NPTS), 256, 0, stream>>>(S0, S2);
    k_centerobj<<<grd(BN_), 256, 0, stream>>>(S2, xyz, o_center, o_obj);

    // 5) feat_s = [sigmoid(mask); feat] ; pre_mlp: 129->128 bn, 128->128
    k_feats<<<grd((long)BATCH * 129 * NPTS), 256, 0, stream>>>(o_mask, S0, S1);
    run_gemm<8, true >(S1, in[26], in[27], in[28], in[29], S2, 129, 128, NPTS, stream);
    run_gemm<8, false>(S2, in[30], in[31], nullptr, nullptr, S1, 128, 128, NPTS, stream); // pre_feat

    // 6) FPS on center_pred, objectness score gather, sample grid, KNN
    k_fps<<<BATCH, 256, 0, stream>>>(o_center, o_propxyz, FPSIDX, o_obj, SCORE);
    k_sample<<<grd((long)BATCH * MROWS), 256, 0, stream>>>(o_propxyz, lwh, SAMPLE);
    k_knn<<<grd((long)BATCH * MROWS), 256, 0, stream>>>(SAMPLE, xyz, KNN);

    // 7) edge features + shared_mlp (134->128 bn, 128->128 bn) over N=4096
    k_edge<<<grd((long)BATCH * NEDGE), 256, 0, stream>>>(KNN, xyz, SAMPLE, S1, E0);
    run_gemm<8, true>(E0, in[32], in[33], in[34], in[35], E1, 134, 128, NEDGE, stream);
    run_gemm<8, true>(E1, in[36], in[37], in[38], in[39], E0, 128, 128, NEDGE, stream);

    // 8) maxpool over K + psf reshape; aggr_conv 1024->128 bn over N=64
    k_maxpsf<<<grd((long)BATCH * 128 * MROWS), 256, 0, stream>>>(E0, PSF);
    run_gemm<8, true>(PSF, in[40], in[41], in[42], in[43], PF0, 1024, 128, PPROP, stream);

    // 9) + center_emb(score); fc_proposal: 128->128 bn, 128->128 bn, 128->5
    k_ceadd<<<grd((long)BATCH * 128 * PPROP), 256, 0, stream>>>(PF0, in[24], in[25], SCORE);
    run_gemm<8, true >(PF0, in[44], in[45], in[46], in[47], PF1, 128, 128, PPROP, stream);
    run_gemm<8, true >(PF1, in[48], in[49], in[50], in[51], PF0, 128, 128, PPROP, stream);
    run_gemm<1, false>(PF0, in[52], in[53], nullptr, nullptr, PF2, 128, 5, PPROP, stream);

    // 10) bboxes
    k_bbox<<<grd((long)BATCH * PPROP), 256, 0, stream>>>(PF2, o_propxyz, o_bbox);

    (void)in_sizes; (void)n_in; (void)out_size; (void)ws_size;
}
